// Attention_82051055223090
// MI455X (gfx1250) — compile-verified, matrix operand order checked by reading, not run on hardware
//
#include <hip/hip_runtime.h>
#include <hip/hip_bf16.h>

typedef __attribute__((ext_vector_type(16))) _Float16 v16h;
typedef __attribute__((ext_vector_type(8)))  _Float16 v8h;
typedef __attribute__((ext_vector_type(8)))  float    v8f;

#define MROWS 18464   // B*N = 32*577
#define CDIM  768
#define NQKV  2304
#define HEADS 12
#define HD    64
#define NTOK  577
#define NPAD  608     // tokens padded to 19*32 for the 32-key attention steps
#define QTILES 37     // ceil(577/16)
#define QGRP   10     // ceil(37/4) q-tiles handled per block (4 waves)

// Build a 16x32 f16 A/B fragment for v_wmma_f32_16x16x32_f16.
// Per ISA layout: lane l holds row (l&15); element e holds K = k0+e (e<8) or
// k0+16+(e-8), with k0 = (l<16) ? 0 : 8.  Both runs are 8 contiguous halfs.
__device__ __forceinline__ v16h make_frag(const _Float16* p0, const _Float16* p1) {
    v8h lo = *(const v8h*)p0;
    v8h hi = *(const v8h*)p1;
    v16h r;
#pragma unroll
    for (int e = 0; e < 8; ++e) { r[e] = lo[e]; r[e + 8] = hi[e]; }
    return r;
}

// CDNA5 async global->LDS DMA (16B per lane), tracked by ASYNCcnt.
// vdst VGPR carries the LDS byte offset (low 32 bits of the generic address).
__device__ __forceinline__ void async_load_b128(unsigned lds_off, const void* gptr) {
    asm volatile("global_load_async_to_lds_b128 %0, %1, off"
                 :: "v"(lds_off), "v"(gptr) : "memory");
}
__device__ __forceinline__ void wait_async0() {
    asm volatile("s_wait_asynccnt 0" ::: "memory");
}

__global__ void f32_to_f16_kernel(const float* __restrict__ in,
                                  _Float16* __restrict__ out, int n) {
    int i = blockIdx.x * 256 + threadIdx.x;
    if (i < n) out[i] = (_Float16)in[i];
}

// ---------------------------------------------------------------------------
// QKV GEMM (64x64 tile / block of 4 waves) + bias + RoPE + q-scale + scatter.
// Each 64-col block covers exactly one (q|k|v, head) pair -> full D=64 in regs.
// ---------------------------------------------------------------------------
__global__ __launch_bounds__(128)
void qkv_rope_kernel(const _Float16* __restrict__ xh,
                     const _Float16* __restrict__ wh,
                     const float* __restrict__ bias,
                     const float* __restrict__ rsin,
                     const float* __restrict__ rcos,
                     _Float16* __restrict__ qb,
                     _Float16* __restrict__ kb,
                     _Float16* __restrict__ vt) {
    __shared__ __align__(16) _Float16 Ash[64 * 32];
    __shared__ __align__(16) _Float16 Bsh[64 * 32];   // stored [n][k]

    const int tid  = threadIdx.x;
    const int lane = tid & 31;
    const int w    = tid >> 5;
    const int nb   = blockIdx.x % 36;   // 2304/64
    const int mb   = blockIdx.x / 36;   // 0..288 (last partial)
    const int mBase = mb * 64, nBase = nb * 64;
    const int l15 = lane & 15, halfSel = lane >> 4;
    const int k0  = (lane < 16) ? 0 : 8;
    const int mloc = 16 * w + l15;

    v8f acc[4] = {};

    for (int kk = 0; kk < CDIM; kk += 32) {
        // Stage A via async DMA: 64x32 halfs as 256 16-byte chunks (2/thread).
        // EXEC-masked lanes (rows >= MROWS) skip; stale LDS rows only feed
        // accumulator rows that the epilogue never stores.
#pragma unroll
        for (int i = 0; i < 2; ++i) {
            int c = tid + 128 * i;
            int row = c >> 2, seg = c & 3;
            int grow = mBase + row;
            unsigned loff = (unsigned)(uintptr_t)(Ash + row * 32 + seg * 8);
            if (grow < MROWS)
                async_load_b128(loff, xh + (size_t)grow * CDIM + kk + seg * 8);
        }
        // Stage B transposed to [n][k]: vector b128 read along n, scatter to LDS.
#pragma unroll
        for (int i = 0; i < 2; ++i) {
            int c = tid + 128 * i;          // 256 chunks of 8 halfs
            int kq = c >> 3;                // 0..31
            int n0 = (c & 7) * 8;
            v8h wv = *(const v8h*)(wh + (size_t)(kk + kq) * NQKV + nBase + n0);
#pragma unroll
            for (int j = 0; j < 8; ++j)
                Bsh[(n0 + j) * 32 + kq] = wv[j];
        }
        // prefetch next K-step panels into L2
        if (kk + 32 < CDIM) {
            int grow = mBase + (tid >> 1);
            if (grow < MROWS)
                __builtin_prefetch(xh + (size_t)grow * CDIM + kk + 32, 0, 1);
            __builtin_prefetch(wh + (size_t)(kk + 32 + (tid & 31)) * NQKV + nBase, 0, 1);
        }
        wait_async0();
        __syncthreads();
        v16h a = make_frag(&Ash[mloc * 32 + k0], &Ash[mloc * 32 + k0 + 16]);
#pragma unroll
        for (int t = 0; t < 4; ++t) {
            int n = 16 * t + l15;
            v16h b = make_frag(&Bsh[n * 32 + k0], &Bsh[n * 32 + k0 + 16]);
            acc[t] = __builtin_amdgcn_wmma_f32_16x16x32_f16(
                false, a, false, b, (short)0, acc[t], false, false);
        }
        __syncthreads();
    }

    const int which = nb / HEADS;        // 0=q 1=k 2=v
    const int h     = nb % HEADS;
#pragma unroll
    for (int t = 0; t < 4; ++t) {
        float bv = bias[nBase + 16 * t + l15];
#pragma unroll
        for (int r = 0; r < 8; ++r) acc[t][r] += bv;
    }
#pragma unroll
    for (int r = 0; r < 8; ++r) {
        int grow = mBase + 16 * w + r + 8 * halfSel;
        if (grow >= MROWS) continue;
        int b   = grow / NTOK;
        int tok = grow % NTOK;
        if (which < 2 && tok > 0) {       // RoPE on non-CLS q/k
#pragma unroll
            for (int t = 0; t < 2; ++t) {
                int d = 16 * t + l15;                  // d < 32; pairs d+32
                float c = rcos[(size_t)(tok - 1) * HD + d];
                float s = rsin[(size_t)(tok - 1) * HD + d];
                float lo = acc[t][r], hi = acc[t + 2][r];
                acc[t][r]     = lo * c - hi * s;
                acc[t + 2][r] = hi * c + lo * s;
            }
        }
        const float qs = (which == 0) ? 0.125f : 1.0f;   // 1/sqrt(64)
        const int bh = b * HEADS + h;
#pragma unroll
        for (int t = 0; t < 4; ++t) {
            int d = 16 * t + l15;
            _Float16 hv = (_Float16)(acc[t][r] * qs);
            if (which == 0)      qb[((size_t)bh * NPAD + tok) * HD + d] = hv;
            else if (which == 1) kb[((size_t)bh * NPAD + tok) * HD + d] = hv;
            else                 vt[((size_t)bh * HD + d) * NPAD + tok] = hv;
        }
    }
}

// ---------------------------------------------------------------------------
// Flash attention: one wave per 16-row q tile; 19 steps of 32 keys.
// ---------------------------------------------------------------------------
__global__ __launch_bounds__(128)
void attn_kernel(const _Float16* __restrict__ qb,
                 const _Float16* __restrict__ kbuf,
                 const _Float16* __restrict__ vt,
                 _Float16* __restrict__ ctx) {
    __shared__ __align__(16) _Float16 pst[4][16 * 32];  // per-wave P staging

    const int tid  = threadIdx.x;
    const int lane = tid & 31;
    const int w    = tid >> 5;
    const int bh   = blockIdx.x / QGRP;
    const int qt   = (blockIdx.x % QGRP) * 4 + w;
    if (qt >= QTILES) return;           // no barriers below -> safe exit
    const int qbase = qt * 16;
    const int l15 = lane & 15, halfSel = lane >> 4;
    const int k0  = (lane < 16) ? 0 : 8;

    const _Float16* qrow = qb + ((size_t)bh * NPAD + qbase + l15) * HD;
    v16h aq0 = make_frag(qrow + k0,      qrow + k0 + 16);       // d 0..31
    v16h aq1 = make_frag(qrow + 32 + k0, qrow + 32 + k0 + 16);  // d 32..63

    float mrow[8], lrow[8];
#pragma unroll
    for (int r = 0; r < 8; ++r) { mrow[r] = -1e30f; lrow[r] = 0.0f; }
    v8f o[4] = {};

    for (int kbs = 0; kbs < NPAD; kbs += 32) {
        if (kbs + 32 < NPAD) {          // prefetch next key/value step
            __builtin_prefetch(kbuf + ((size_t)bh * NPAD + kbs + 32 + lane) * HD, 0, 1);
            __builtin_prefetch(vt + ((size_t)bh * HD + lane * 2) * NPAD + kbs + 32, 0, 1);
        }
        // --- scores: S = Q K^T, two 16-key sub-tiles ---
        v8f s0 = {}, s1 = {};
        {
            const _Float16* kr = kbuf + ((size_t)bh * NPAD + kbs + l15) * HD;
            v16h b0 = make_frag(kr + k0,      kr + k0 + 16);
            v16h b1 = make_frag(kr + 32 + k0, kr + 32 + k0 + 16);
            s0 = __builtin_amdgcn_wmma_f32_16x16x32_f16(false, aq0, false, b0, (short)0, s0, false, false);
            s0 = __builtin_amdgcn_wmma_f32_16x16x32_f16(false, aq1, false, b1, (short)0, s0, false, false);
        }
        {
            const _Float16* kr = kbuf + ((size_t)bh * NPAD + kbs + 16 + l15) * HD;
            v16h b0 = make_frag(kr + k0,      kr + k0 + 16);
            v16h b1 = make_frag(kr + 32 + k0, kr + 32 + k0 + 16);
            s1 = __builtin_amdgcn_wmma_f32_16x16x32_f16(false, aq0, false, b0, (short)0, s1, false, false);
            s1 = __builtin_amdgcn_wmma_f32_16x16x32_f16(false, aq1, false, b1, (short)0, s1, false, false);
        }
        // mask out-of-range key tokens
        if (kbs + l15 >= NTOK) {
#pragma unroll
            for (int r = 0; r < 8; ++r) s0[r] = -1e30f;
        }
        if (kbs + 16 + l15 >= NTOK) {
#pragma unroll
            for (int r = 0; r < 8; ++r) s1[r] = -1e30f;
        }
        // --- online softmax (row = 16 lanes of a half-wave, per reg r) ---
        float p0[8], p1[8];
#pragma unroll
        for (int r = 0; r < 8; ++r) {
            float mx = fmaxf(s0[r], s1[r]);
            for (int off = 1; off < 16; off <<= 1)
                mx = fmaxf(mx, __shfl_xor(mx, off, 32));
            float mnew  = fmaxf(mrow[r], mx);
            float scale = __expf(mrow[r] - mnew);
            float a0 = __expf(s0[r] - mnew);
            float a1 = __expf(s1[r] - mnew);
            float sum = a0 + a1;
            for (int off = 1; off < 16; off <<= 1)
                sum += __shfl_xor(sum, off, 32);
            lrow[r] = lrow[r] * scale + sum;
            mrow[r] = mnew;
            p0[r] = a0; p1[r] = a1;
#pragma unroll
            for (int t = 0; t < 4; ++t) o[t][r] *= scale;
        }
        // --- P: D-layout -> A-layout via per-wave LDS slice ---
        _Float16* ps = &pst[w][0];
#pragma unroll
        for (int r = 0; r < 8; ++r) {
            int row = r + 8 * halfSel;
            ps[row * 32 + l15]      = (_Float16)p0[r];
            ps[row * 32 + 16 + l15] = (_Float16)p1[r];
        }
        asm volatile("s_wait_dscnt 0" ::: "memory");  // DS in-order per wave
        v16h ap = make_frag(ps + l15 * 32 + k0, ps + l15 * 32 + k0 + 16);
        // --- O += P V (V stored d-major so k-runs are contiguous) ---
#pragma unroll
        for (int t = 0; t < 4; ++t) {
            const _Float16* vr = vt + ((size_t)bh * HD + 16 * t + l15) * NPAD + kbs;
            v16h bv = make_frag(vr + k0, vr + k0 + 16);
            o[t] = __builtin_amdgcn_wmma_f32_16x16x32_f16(
                false, ap, false, bv, (short)0, o[t], false, false);
        }
    }
    // --- normalize + store ctx[B,N,C] ---
    const int b = bh / HEADS, h = bh % HEADS;
#pragma unroll
    for (int r = 0; r < 8; ++r) {
        int tok = qbase + r + 8 * halfSel;
        if (tok >= NTOK) continue;
        float inv = 1.0f / lrow[r];
#pragma unroll
        for (int t = 0; t < 4; ++t) {
            int d = 16 * t + l15;
            ctx[((size_t)(b * NTOK + tok)) * CDIM + h * HD + d] =
                (_Float16)(o[t][r] * inv);
        }
    }
}

// ---------------------------------------------------------------------------
// Output projection: out(f32) = ctx(f16) @ w_proj(f16) + b_proj
// ---------------------------------------------------------------------------
__global__ __launch_bounds__(128)
void proj_kernel(const _Float16* __restrict__ ctx,
                 const _Float16* __restrict__ wh,
                 const float* __restrict__ bias,
                 float* __restrict__ out) {
    __shared__ __align__(16) _Float16 Ash[64 * 32];
    __shared__ __align__(16) _Float16 Bsh[64 * 32];

    const int tid  = threadIdx.x;
    const int lane = tid & 31;
    const int w    = tid >> 5;
    const int nb   = blockIdx.x % 12;   // 768/64
    const int mb   = blockIdx.x / 12;
    const int mBase = mb * 64, nBase = nb * 64;
    const int l15 = lane & 15, halfSel = lane >> 4;
    const int k0  = (lane < 16) ? 0 : 8;
    const int mloc = 16 * w + l15;

    v8f acc[4] = {};
    for (int kk = 0; kk < CDIM; kk += 32) {
#pragma unroll
        for (int i = 0; i < 2; ++i) {
            int c = tid + 128 * i;
            int row = c >> 2, seg = c & 3;
            int grow = mBase + row;
            unsigned loff = (unsigned)(uintptr_t)(Ash + row * 32 + seg * 8);
            if (grow < MROWS)
                async_load_b128(loff, ctx + (size_t)grow * CDIM + kk + seg * 8);
        }
#pragma unroll
        for (int i = 0; i < 2; ++i) {
            int c = tid + 128 * i;
            int kq = c >> 3;
            int n0 = (c & 7) * 8;
            v8h wv = *(const v8h*)(wh + (size_t)(kk + kq) * CDIM + nBase + n0);
#pragma unroll
            for (int j = 0; j < 8; ++j)
                Bsh[(n0 + j) * 32 + kq] = wv[j];
        }
        if (kk + 32 < CDIM) {
            int grow = mBase + (tid >> 1);
            if (grow < MROWS)
                __builtin_prefetch(ctx + (size_t)grow * CDIM + kk + 32, 0, 1);
            __builtin_prefetch(wh + (size_t)(kk + 32 + (tid & 31)) * CDIM + nBase, 0, 1);
        }
        wait_async0();
        __syncthreads();
        v16h a = make_frag(&Ash[mloc * 32 + k0], &Ash[mloc * 32 + k0 + 16]);
#pragma unroll
        for (int t = 0; t < 4; ++t) {
            int n = 16 * t + l15;
            v16h b = make_frag(&Bsh[n * 32 + k0], &Bsh[n * 32 + k0 + 16]);
            acc[t] = __builtin_amdgcn_wmma_f32_16x16x32_f16(
                false, a, false, b, (short)0, acc[t], false, false);
        }
        __syncthreads();
    }
#pragma unroll
    for (int r = 0; r < 8; ++r) {
        int grow = mBase + 16 * w + r + 8 * halfSel;
        if (grow >= MROWS) continue;
#pragma unroll
        for (int t = 0; t < 4; ++t) {
            int n = nBase + 16 * t + l15;
            out[(size_t)grow * CDIM + n] = acc[t][r] + bias[n];
        }
    }
}

extern "C" void kernel_launch(void* const* d_in, const int* in_sizes, int n_in,
                              void* d_out, int out_size, void* d_ws, size_t ws_size,
                              hipStream_t stream) {
    const float* x      = (const float*)d_in[0];
    const float* w_qkv  = (const float*)d_in[1];
    const float* b_qkv  = (const float*)d_in[2];
    const float* w_proj = (const float*)d_in[3];
    const float* b_proj = (const float*)d_in[4];
    const float* rsin   = (const float*)d_in[5];
    const float* rcos   = (const float*)d_in[6];
    float* out = (float*)d_out;

    const size_t XH_B   = (size_t)MROWS * CDIM * 2;   // 28,360,704
    const size_t WQ_B   = (size_t)CDIM * NQKV * 2;    //  3,538,944
    const size_t WP_B   = (size_t)CDIM * CDIM * 2;    //  1,179,648
    const size_t PAD_B  = (size_t)32 * HEADS * NPAD * HD * 2; // 29,884,416

    char* p = (char*)d_ws;
    _Float16* xh   = (_Float16*)p; p += XH_B;
    _Float16* wqh  = (_Float16*)p; p += WQ_B;
    _Float16* wph  = (_Float16*)p; p += WP_B;
    _Float16* qbuf = (_Float16*)p; p += PAD_B;
    _Float16* kbuf = (_Float16*)p; p += PAD_B;
    _Float16* vtb  = (_Float16*)p; p += PAD_B;
    _Float16* ctx  = (_Float16*)p; p += XH_B;

    // zero padded token regions of q/k/v^T (contiguous -> single memset node)
    hipMemsetAsync(qbuf, 0, 3 * PAD_B, stream);

    int nx = MROWS * CDIM, nwq = CDIM * NQKV, nwp = CDIM * CDIM;
    f32_to_f16_kernel<<<(nx  + 255) / 256, 256, 0, stream>>>(x,      xh,  nx);
    f32_to_f16_kernel<<<(nwq + 255) / 256, 256, 0, stream>>>(w_qkv,  wqh, nwq);
    f32_to_f16_kernel<<<(nwp + 255) / 256, 256, 0, stream>>>(w_proj, wph, nwp);

    qkv_rope_kernel<<<289 * 36, 128, 0, stream>>>(xh, wqh, b_qkv, rsin, rcos,
                                                  qbuf, kbuf, vtb);
    attn_kernel<<<32 * HEADS * QGRP, 128, 0, stream>>>(qbuf, kbuf, vtb, ctx);
    proj_kernel<<<289 * 12, 128, 0, stream>>>(ctx, wph, b_proj, out);
}
